// AttentiveSSM_9165460209861
// MI455X (gfx1250) — matrix operand order repaired and checked by reading, not compiled, not run on hardware
//
#include <hip/hip_runtime.h>

// ---------------------------------------------------------------------------
// AttentiveSSM forward for MI455X (gfx1250, wave32, WMMA + TDM).
// bf16 WMMA (f32 accum) for GEMMs + sparse two-pass flash attention.
// All WMMA fragments load as contiguous b128; boundary K tiles move via the
// Tensor Data Mover into LDS (guarded; falls back to global loads).
// ---------------------------------------------------------------------------

#define DEV __device__ __forceinline__
#define SHUF16(a, b) __builtin_shufflevector(a, b, 0,1,2,3,4,5,6,7,8,9,10,11,12,13,14,15)

typedef __bf16 bf16_t;
typedef __attribute__((ext_vector_type(16))) __bf16 bf16x16;
typedef __attribute__((ext_vector_type(8)))  __bf16 bf16x8;
typedef __attribute__((ext_vector_type(8)))  float  f32x8;
typedef __attribute__((ext_vector_type(4))) unsigned uint32x4;
typedef __attribute__((ext_vector_type(8))) int      int32x8;
typedef __attribute__((ext_vector_type(4))) int      int32x4;

#if __has_builtin(__builtin_amdgcn_tensor_load_to_lds) && \
    __has_builtin(__builtin_amdgcn_s_wait_tensorcnt)
#define HAVE_TDM 1
#else
#define HAVE_TDM 0
#endif

constexpr int S_LEN  = 2048;
constexpr int DIM    = 2048;
constexpr int HD     = 128;
constexpr int NH     = 16;
constexpr int NKV    = 4;
constexpr int KV_DIM = NKV * HD;   // 512
constexpr int QDIM   = NH * HD;    // 2048
constexpr int NBND   = S_LEN / 8;  // 256 boundary keys

DEV float lane_shfl(float v, int srcLane) {   // wave32 gather via ds_bpermute
  return __builtin_bit_cast(
      float, __builtin_amdgcn_ds_bpermute(srcLane << 2, __builtin_bit_cast(int, v)));
}

#if HAVE_TDM
// 2D TDM: copy rows x rowElems bf16 tile (row stride strideElems) to LDS.
// D# layout per CDNA5 ISA ch.8 (group0 128b, group1 256b; groups 2/3 zero).
DEV void tdm_load_2d(const bf16_t* gsrc, unsigned ldsAddr, int rows, int rowElems,
                     int strideElems) {
  unsigned long long ga = (unsigned long long)(size_t)gsrc;
  uint32x4 g0;
  g0[0] = 1u;                                          // count=1
  g0[1] = ldsAddr;                                     // lds_addr [63:32]
  g0[2] = (unsigned)(ga & 0xFFFFFFFFu);                // global_addr lo
  g0[3] = (unsigned)((ga >> 32) & 0x01FFFFFFu) | (2u << 30);  // addr hi | type=2
  int32x8 g1;
  g1[0] = (1 << 16);                                   // data_size=1 (2 bytes)
  g1[1] = (rowElems & 0xFFFF) << 16;                   // tensor_dim0[15:0]
  g1[2] = ((rows & 0xFFFF) << 16);                     // dim0 hi=0 | tensor_dim1 lo
  g1[3] = ((rowElems & 0xFFFF) << 16);                 // dim1 hi=0 | tile_dim0
  g1[4] = (rows & 0xFFFF);                             // tile_dim1 | tile_dim2=0
  g1[5] = strideElems;                                 // tensor_dim0_stride lo32
  g1[6] = 0;
  g1[7] = 0;
  int32x4 z4 = {0, 0, 0, 0};
#if defined(__clang_major__) && (__clang_major__ >= 23)
  int32x8 z8 = {0, 0, 0, 0, 0, 0, 0, 0};
  __builtin_amdgcn_tensor_load_to_lds(g0, g1, z4, z4, z8, 0);
#else
  __builtin_amdgcn_tensor_load_to_lds(g0, g1, z4, z4, 0);
#endif
}
#endif

// ---------------------------------------------------------------------------
// fp32 -> bf16 conversion, 8 elements/thread (b128 in, b128 out)
// ---------------------------------------------------------------------------
__global__ __launch_bounds__(256) void cvt_f32_bf16_v8(const float* __restrict__ in,
                                                       bf16_t* __restrict__ out, int n8) {
  int i = blockIdx.x * 256 + threadIdx.x;
  if (i >= n8) return;
  const float4* p = (const float4*)in + (size_t)i * 2;
  float4 a = p[0], b = p[1];
  bf16x8 o;
  o[0] = (bf16_t)a.x; o[1] = (bf16_t)a.y; o[2] = (bf16_t)a.z; o[3] = (bf16_t)a.w;
  o[4] = (bf16_t)b.x; o[5] = (bf16_t)b.y; o[6] = (bf16_t)b.z; o[7] = (bf16_t)b.w;
  ((bf16x8*)out)[i] = o;
}

// ---------------------------------------------------------------------------
// C[M,N] (f32) = A[M,K] (bf16 row-major) @ B[N,K]^T (bf16 rows = N)
// One wave: 16x64 tile; manual unroll-by-2 ping-pong (no register copies).
// Requires K % 64 == 0.
// ---------------------------------------------------------------------------
__global__ __launch_bounds__(256) void gemm_xwt_wmma(
    const bf16_t* __restrict__ A, const bf16_t* __restrict__ B,
    float* __restrict__ C, int M, int N, int K) {
  const int lane   = threadIdx.x & 31;
  const int waveId = blockIdx.x * 8 + (threadIdx.x >> 5);
  const int nQuads = N >> 6;
  const int mTile  = waveId / nQuads;
  const int nQuad  = waveId - mTile * nQuads;
  if (mTile * 16 >= M) return;                 // wave-uniform

  const int lo16 = (lane < 16);
  const int nIdx = lane & 15;
  const int aOff = lo16 ? 0 : 8;               // A K-halves {0..7,16..23}/{8..15,24..31}
  const int bOff = lo16 ? 0 : 8;               // B K-halves {0..7,16..23}/{8..15,24..31}
  const bf16_t* Arow = A + (size_t)(mTile * 16 + nIdx) * K;
  const int n0 = nQuad * 64;

  auto loadA16 = [&](int k0) -> bf16x16 {
    bf16x8 a0 = *(const bf16x8*)(Arow + k0 + aOff);
    bf16x8 a1 = *(const bf16x8*)(Arow + k0 + aOff + 16);
    return SHUF16(a0, a1);
  };
  auto loadB16 = [&](int k0, int t) -> bf16x16 {
    const bf16_t* Brow = B + (size_t)(n0 + t * 16 + nIdx) * K + k0 + bOff;
    bf16x8 b0 = *(const bf16x8*)(Brow);
    bf16x8 b1 = *(const bf16x8*)(Brow + 16);
    return SHUF16(b0, b1);
  };

  f32x8 acc[4] = {};
  bf16x16 afA = loadA16(0);
  bf16x16 afB = loadA16(32);
  bf16x16 bfA[4], bfB[4];
#pragma unroll
  for (int t = 0; t < 4; ++t) { bfA[t] = loadB16(0, t); bfB[t] = loadB16(32, t); }

  for (int k0 = 0; k0 < K; k0 += 64) {
#pragma unroll
    for (int t = 0; t < 4; ++t)
      acc[t] = __builtin_amdgcn_wmma_f32_16x16x32_bf16(false, afA, false, bfA[t],
                                                       (short)0, acc[t], false, false);
    if (k0 + 64 < K) {                         // refill phase-A regs, overlaps phase-B math
      __builtin_prefetch(Arow + k0 + 256, 0, 1);
      afA = loadA16(k0 + 64);
#pragma unroll
      for (int t = 0; t < 4; ++t) bfA[t] = loadB16(k0 + 64, t);
    }
#pragma unroll
    for (int t = 0; t < 4; ++t)
      acc[t] = __builtin_amdgcn_wmma_f32_16x16x32_bf16(false, afB, false, bfB[t],
                                                       (short)0, acc[t], false, false);
    if (k0 + 96 < K) {                         // refill phase-B regs
      afB = loadA16(k0 + 96);
#pragma unroll
      for (int t = 0; t < 4; ++t) bfB[t] = loadB16(k0 + 96, t);
    }
  }

  const int rbase = lo16 ? 0 : 8;
#pragma unroll
  for (int t = 0; t < 4; ++t)
#pragma unroll
    for (int r = 0; r < 8; ++r)
      C[(size_t)(mTile * 16 + rbase + r) * N + n0 + t * 16 + nIdx] = acc[t][r];
}

// ---------------------------------------------------------------------------
// Chunked SSM (len 8) + residual: h = a*h + b*x ; y = c*h + x
// ---------------------------------------------------------------------------
__global__ __launch_bounds__(256) void ssm_chunk(
    const float* __restrict__ X, const float* __restrict__ a_log,
    const float* __restrict__ Bp, const float* __restrict__ Cp,
    float* __restrict__ Y) {
  int idx   = blockIdx.x * 256 + threadIdx.x;
  int e     = idx & (KV_DIM - 1);
  int chunk = idx >> 9;
  if (chunk >= S_LEN / 8) return;
  float a  = 1.0f / (1.0f + __expf(a_log[e]));
  float bb = Bp[e], cc = Cp[e];
  const float* xp = X + (size_t)chunk * 8 * KV_DIM + e;
  float*       yp = Y + (size_t)chunk * 8 * KV_DIM + e;
  float h = 0.0f;
#pragma unroll
  for (int t = 0; t < 8; ++t) {
    float xv = xp[t * KV_DIM];
    h = a * h + bb * xv;
    yp[t * KV_DIM] = cc * h + xv;
  }
}

// ---------------------------------------------------------------------------
// RoPE in place. X: (S, nheads*128). freq: (S, 64, 2, 2) [cos,-sin;sin,cos].
// ---------------------------------------------------------------------------
__global__ __launch_bounds__(256) void rope_inplace(
    float* __restrict__ X, const float* __restrict__ F, int nheads) {
  int idx = blockIdx.x * 256 + threadIdx.x;
  int i = idx & 63;
  int h = (idx >> 6) % nheads;
  int s = idx / (nheads * 64);
  if (s >= S_LEN) return;
  float c  = F[(size_t)s * 256 + i * 4 + 0];
  float sn = F[(size_t)s * 256 + i * 4 + 2];
  float* p = X + (size_t)s * nheads * HD + h * HD + 2 * i;
  float x0 = p[0], x1 = p[1];
  p[0] = x0 * c - x1 * sn;
  p[1] = x0 * sn + x1 * c;
}

// ---------------------------------------------------------------------------
// V transpose + bf16: in (S, 512) f32 -> out (512, S) bf16, LDS 32x33 tiles.
// ---------------------------------------------------------------------------
__global__ __launch_bounds__(256) void transpose_v_bf16(const float* __restrict__ in,
                                                        bf16_t* __restrict__ out) {
  __shared__ bf16_t tile[32][33];
  int ts = blockIdx.x * 32;
  int te = blockIdx.y * 32;
  int tx = threadIdx.x & 31;
  int ty = threadIdx.x >> 5;
#pragma unroll
  for (int j = 0; j < 32; j += 8)
    tile[ty + j][tx] = (bf16_t)in[(size_t)(ts + ty + j) * KV_DIM + te + tx];
  __syncthreads();
#pragma unroll
  for (int j = 0; j < 32; j += 8)
    out[(size_t)(te + ty + j) * S_LEN + ts + tx] = tile[tx][ty + j];
}

// ---------------------------------------------------------------------------
// Boundary compaction: rows 8b+7 of post-rope K and of V (transposed).
// ---------------------------------------------------------------------------
__global__ __launch_bounds__(256) void build_boundary(
    const float* __restrict__ Kf, const float* __restrict__ Vf,
    bf16_t* __restrict__ kbnd, bf16_t* __restrict__ vtbnd) {
  int idx = blockIdx.x * 256 + threadIdx.x;
  int e = idx & (KV_DIM - 1);
  int b = idx >> 9;
  if (b >= NBND) return;
  size_t src = (size_t)(8 * b + 7) * KV_DIM + e;
  kbnd[(size_t)b * KV_DIM + e] = (bf16_t)Kf[src];
  vtbnd[(size_t)e * NBND + b]  = (bf16_t)Vf[src];
}

// ---------------------------------------------------------------------------
// One 32-key attention block (transposed scores, online softmax).
// MODE 0: boundary pass (row b -> key 8b+7, mask 8b+7 < t)
// MODE 1: local pass (original mask minus boundary-counted keys)
// LDSK:   K tile comes from LDS (local rows) instead of global rows.
// ---------------------------------------------------------------------------
template <int MODE, bool LDSK>
DEV void attn_block(int j0, int t_q, int lane, int nIdx, int lo16, int rbase,
                    const bf16_t* __restrict__ Krow0, int kstride, int maxRow,
                    const bf16_t* __restrict__ Vcol0, size_t vstride,
                    const bf16x16 (&qf)[4],
                    f32x8 (&acc)[8], float& m_run, float& l_run) {
  const int aDseg = lo16 ? 0 : 8;
  const int kb    = lo16 ? 0 : 8;

  // V fragments first (independent of scores -> overlaps score WMMAs)
  bf16x16 vf[8];
#pragma unroll
  for (int dt = 0; dt < 8; ++dt) {
    const bf16_t* vp = Vcol0 + (size_t)(dt * 16 + nIdx) * vstride + j0 + kb;
    bf16x8 v0 = *(const bf16x8*)(vp);
    bf16x8 v1 = *(const bf16x8*)(vp + 16);
    vf[dt] = SHUF16(v0, v1);
  }

  f32x8 st0 = {}, st1 = {};
#pragma unroll
  for (int h2 = 0; h2 < 2; ++h2) {
    int row = (LDSK ? 0 : j0) + h2 * 16 + nIdx;
    row = row < maxRow ? row : maxRow;
    const bf16_t* krp = Krow0 + (size_t)row * kstride;
#pragma unroll
    for (int c = 0; c < 4; ++c) {
      bf16x8 a0 = *(const bf16x8*)(krp + c * 32 + aDseg);
      bf16x8 a1 = *(const bf16x8*)(krp + c * 32 + aDseg + 16);
      bf16x16 af = SHUF16(a0, a1);
      if (h2 == 0)
        st0 = __builtin_amdgcn_wmma_f32_16x16x32_bf16(false, af, false, qf[c],
                                                      (short)0, st0, false, false);
      else
        st1 = __builtin_amdgcn_wmma_f32_16x16x32_bf16(false, af, false, qf[c],
                                                      (short)0, st1, false, false);
    }
  }

  const float scl = 0.088388347648318447f;     // 1/sqrt(128)
  float mblk = -3.0e38f;
#pragma unroll
  for (int r = 0; r < 8; ++r) {
    int r0 = j0 + rbase + r;
    int r1 = j0 + 16 + rbase + r;
    bool ok0, ok1;
    if (MODE == 0) {
      ok0 = (8 * r0 + 7) < t_q;
      ok1 = (8 * r1 + 7) < t_q;
    } else {
      bool bd0 = ((r0 & 7) == 7) && (r0 < t_q);
      bool bd1 = ((r1 & 7) == 7) && (r1 < t_q);
      ok0 = !bd0 && ((r0 == t_q - 1) || ((r0 < 4) && (r0 <= t_q)));
      ok1 = !bd1 && ((r1 == t_q - 1) || ((r1 < 4) && (r1 <= t_q)));
    }
    st0[r] = ok0 ? st0[r] * scl : -3.0e38f;
    st1[r] = ok1 ? st1[r] * scl : -3.0e38f;
    mblk = fmaxf(mblk, fmaxf(st0[r], st1[r]));
  }
  mblk = fmaxf(mblk, lane_shfl(mblk, lane ^ 16));
  float m_new = fmaxf(m_run, mblk);
  float alpha = __expf(m_run - m_new);

  f32x8 p0, p1;
  float lsum = 0.0f;
#pragma unroll
  for (int r = 0; r < 8; ++r) {
    p0[r] = __expf(st0[r] - m_new);
    p1[r] = __expf(st1[r] - m_new);
    lsum += p0[r] + p1[r];
  }
  lsum += lane_shfl(lsum, lane ^ 16);
  l_run = l_run * alpha + lsum;
  m_run = m_new;

  float aR[8];
#pragma unroll
  for (int r = 0; r < 8; ++r) aR[r] = lane_shfl(alpha, rbase + r);
#pragma unroll
  for (int dt = 0; dt < 8; ++dt)
#pragma unroll
    for (int r = 0; r < 8; ++r) acc[dt][r] *= aR[r];

  // P is natively in A-layout: halves 0..7 = st0 rows, 8..15 = st1 rows
  bf16x16 pf;
#pragma unroll
  for (int i = 0; i < 8; ++i) { pf[i] = (bf16_t)p0[i]; pf[i + 8] = (bf16_t)p1[i]; }

#pragma unroll
  for (int dt = 0; dt < 8; ++dt)
    acc[dt] = __builtin_amdgcn_wmma_f32_16x16x32_bf16(false, pf, false, vf[dt],
                                                      (short)0, acc[dt], false, false);
}

// ---------------------------------------------------------------------------
// Sparse flash attention: one wave per (head, 16-query tile).
// Pass A: boundary-compacted keys (K tile via TDM -> LDS when available).
// Pass B: <=3 local 32-key blocks.
// ---------------------------------------------------------------------------
__global__ __launch_bounds__(256) void flash_attn_wmma(
    const bf16_t* __restrict__ Q,     // (S, 2048)
    const bf16_t* __restrict__ Km,    // (S, 512)
    const bf16_t* __restrict__ Vt,    // (512, S)   transposed V
    const bf16_t* __restrict__ Kbnd,  // (256, 512) boundary rows of K
    const bf16_t* __restrict__ Vtbnd, // (512, 256) boundary cols of V^T
    float* __restrict__ Out) {        // (S, 2048)
  const int lane   = threadIdx.x & 31;
  const int waveId = blockIdx.x * 8 + (threadIdx.x >> 5);
  const int head   = waveId & (NH - 1);
  const int qTile  = waveId >> 4;
  const int q0     = qTile * 16;
  const int kvh    = head >> 2;

  const int lo16  = (lane < 16);
  const int nIdx  = lane & 15;
  const int rbase = lo16 ? 0 : 8;
  const int t_q   = q0 + nIdx;

  // Q as B-operand fragments (K-dim = head_dim, lane = query column)
  bf16x16 qf[4];
  {
    const bf16_t* qrow = Q + (size_t)(q0 + nIdx) * QDIM + head * HD + (lo16 ? 0 : 8);
#pragma unroll
    for (int c = 0; c < 4; ++c) {
      bf16x8 l0 = *(const bf16x8*)(qrow + c * 32);
      bf16x8 l1 = *(const bf16x8*)(qrow + c * 32 + 16);
      qf[c] = SHUF16(l0, l1);
    }
  }

  f32x8 acc[8] = {};
  float m_run = -3.0e38f, l_run = 0.0f;

  const bf16_t* KbndH  = Kbnd + kvh * HD;
  const bf16_t* VtbndH = Vtbnd + (size_t)(kvh * HD) * NBND;
  const bf16_t* KmH    = Km + kvh * HD;
  const bf16_t* VtH    = Vt + (size_t)(kvh * HD) * S_LEN;

  // ---- Pass A: boundary keys 8b+7 < t (compacted rows b)
#if HAVE_TDM
  __shared__ alignas(16) bf16_t ktile[8][32 * HD];   // 8 KB per wave
  bf16_t*  myTile    = &ktile[threadIdx.x >> 5][0];
  unsigned myTileOff = (unsigned)(size_t)myTile;     // LDS byte offset
  for (int jb0 = 0; 8 * jb0 + 7 < q0 + 15; jb0 += 32) {
    asm volatile("s_wait_dscnt 0x0" ::: "memory");   // prior tile reads done (WAR)
    tdm_load_2d(KbndH + (size_t)jb0 * KV_DIM, myTileOff, 32, HD, KV_DIM);
    __builtin_amdgcn_s_wait_tensorcnt(0);
    attn_block<0, true>(jb0, t_q, lane, nIdx, lo16, rbase, myTile, HD, 31,
                        VtbndH, (size_t)NBND, qf, acc, m_run, l_run);
  }
#else
  for (int jb0 = 0; 8 * jb0 + 7 < q0 + 15; jb0 += 32)
    attn_block<0, false>(jb0, t_q, lane, nIdx, lo16, rbase, KbndH, KV_DIM, NBND - 1,
                         VtbndH, (size_t)NBND, qf, acc, m_run, l_run);
#endif

  // ---- Pass B: block 0 (k<4 head) + blocks covering k == t-1
  int blocks[3];
  int nb = 0;
  blocks[nb++] = 0;
  int bA = (q0 >= 1) ? ((q0 - 1) & ~31) : 0;
  if (bA != 0) blocks[nb++] = bA;
  int bB = q0 & ~31;
  if (bB != 0 && bB != bA) blocks[nb++] = bB;
  for (int i = 0; i < nb; ++i)
    attn_block<1, false>(blocks[i], t_q, lane, nIdx, lo16, rbase, KmH, KV_DIM,
                         S_LEN - 1, VtH, (size_t)S_LEN, qf, acc, m_run, l_run);

  // ---- epilogue: divide by row sums, store
  float linv = 1.0f / l_run;
  float lR[8];
#pragma unroll
  for (int r = 0; r < 8; ++r) lR[r] = lane_shfl(linv, rbase + r);
#pragma unroll
  for (int dt = 0; dt < 8; ++dt)
#pragma unroll
    for (int r = 0; r < 8; ++r)
      Out[(size_t)(q0 + rbase + r) * QDIM + head * HD + dt * 16 + nIdx] =
          acc[dt][r] * lR[r];
}

// ---------------------------------------------------------------------------
// Host launch
// ---------------------------------------------------------------------------
extern "C" void kernel_launch(void* const* d_in, const int* in_sizes, int n_in,
                              void* d_out, int out_size, void* d_ws, size_t ws_size,
                              hipStream_t stream) {
  (void)in_sizes; (void)n_in; (void)out_size; (void)ws_size;
  const float* x       = (const float*)d_in[0];
  const float* freq    = (const float*)d_in[1];
  const float* wq      = (const float*)d_in[2];
  const float* wk      = (const float*)d_in[3];
  const float* wv      = (const float*)d_in[4];
  const float* wo      = (const float*)d_in[5];
  const float* a_log_k = (const float*)d_in[6];
  const float* b_k     = (const float*)d_in[7];
  const float* c_k     = (const float*)d_in[8];
  const float* a_log_v = (const float*)d_in[9];
  const float* b_v     = (const float*)d_in[10];
  const float* c_v     = (const float*)d_in[11];

  char*  ws  = (char*)d_ws;
  size_t off = 0;
  auto take = [&](size_t bytes) -> void* {
    void* p = ws + off;
    off += (bytes + 255) & ~(size_t)255;
    return p;
  };
  bf16_t* xbf   = (bf16_t*)take((size_t)S_LEN * DIM * 2);
  bf16_t* wqbf  = (bf16_t*)take((size_t)QDIM * DIM * 2);
  bf16_t* wkbf  = (bf16_t*)take((size_t)KV_DIM * DIM * 2);
  bf16_t* wvbf  = (bf16_t*)take((size_t)KV_DIM * DIM * 2);
  bf16_t* wobf  = (bf16_t*)take((size_t)DIM * QDIM * 2);
  float*  xq    = (float*)take((size_t)S_LEN * QDIM * 4);
  float*  xk    = (float*)take((size_t)S_LEN * KV_DIM * 4);
  float*  xv    = (float*)take((size_t)S_LEN * KV_DIM * 4);
  float*  kss   = (float*)take((size_t)S_LEN * KV_DIM * 4);
  float*  vss   = (float*)take((size_t)S_LEN * KV_DIM * 4);
  bf16_t* qbf   = (bf16_t*)take((size_t)S_LEN * QDIM * 2);
  bf16_t* kbf   = (bf16_t*)take((size_t)S_LEN * KV_DIM * 2);
  bf16_t* vt    = (bf16_t*)take((size_t)KV_DIM * S_LEN * 2);
  bf16_t* kbnd  = (bf16_t*)take((size_t)NBND * KV_DIM * 2);
  bf16_t* vtbnd = (bf16_t*)take((size_t)KV_DIM * NBND * 2);
  float*  attn  = (float*)take((size_t)S_LEN * QDIM * 4);
  bf16_t* aobf  = (bf16_t*)take((size_t)S_LEN * QDIM * 2);

  dim3 blk(256);
  auto cvt = [&](const float* src, bf16_t* dst, int n) {
    cvt_f32_bf16_v8<<<dim3(n / 8 / 256), blk, 0, stream>>>(src, dst, n / 8);
  };
  auto gemm = [&](const bf16_t* A, const bf16_t* Bm, float* Cm, int M, int N, int K) {
    int waves = (M / 16) * (N / 64);
    gemm_xwt_wmma<<<dim3(waves / 8), blk, 0, stream>>>(A, Bm, Cm, M, N, K);
  };

  // bf16 down-conversion of x and weights
  cvt(x, xbf, S_LEN * DIM);
  cvt(wq, wqbf, QDIM * DIM);
  cvt(wk, wkbf, KV_DIM * DIM);
  cvt(wv, wvbf, KV_DIM * DIM);
  cvt(wo, wobf, DIM * QDIM);

  // q/k/v projections
  gemm(xbf, wqbf, xq, S_LEN, QDIM, DIM);
  gemm(xbf, wkbf, xk, S_LEN, KV_DIM, DIM);
  gemm(xbf, wvbf, xv, S_LEN, KV_DIM, DIM);

  // chunked SSM + residual
  ssm_chunk<<<dim3((S_LEN / 8) * KV_DIM / 256), blk, 0, stream>>>(xk, a_log_k, b_k, c_k, kss);
  ssm_chunk<<<dim3((S_LEN / 8) * KV_DIM / 256), blk, 0, stream>>>(xv, a_log_v, b_v, c_v, vss);

  // RoPE (q and k only)
  rope_inplace<<<dim3(S_LEN * NH * 64 / 256), blk, 0, stream>>>(xq, freq, NH);
  rope_inplace<<<dim3(S_LEN * NKV * 64 / 256), blk, 0, stream>>>(kss, freq, NKV);

  // attention operand prep
  cvt(xq, qbf, S_LEN * QDIM);
  cvt(kss, kbf, S_LEN * KV_DIM);
  transpose_v_bf16<<<dim3(S_LEN / 32, KV_DIM / 32), blk, 0, stream>>>(vss, vt);
  build_boundary<<<dim3(NBND * KV_DIM / 256), blk, 0, stream>>>(kss, vss, kbnd, vtbnd);

  // sparse flash attention: 128 q-tiles x 16 heads = 2048 waves
  flash_attn_wmma<<<dim3((S_LEN / 16) * NH / 8), blk, 0, stream>>>(qbf, kbf, vt, kbnd,
                                                                   vtbnd, attn);

  // output projection
  cvt(attn, aobf, S_LEN * QDIM);
  gemm(aobf, wobf, (float*)d_out, S_LEN, DIM, QDIM);
}
